// LocalWindowAttention_57329223467171
// MI455X (gfx1250) — compile-verified, matrix operand order checked by reading, not run on hardware
//
#include <hip/hip_runtime.h>

// ---------------------------------------------------------------------------
// Swin-style local window attention for MI455X (gfx1250, wave32, WMMA).
// bf16 WMMA (v_wmma_f32_16x16x32_bf16) for both the QKV projection GEMM and
// the per-window attention GEMMs; f32 accumulation and f32 softmax.
// ---------------------------------------------------------------------------

typedef __attribute__((ext_vector_type(16))) __bf16 v16bf;
typedef __attribute__((ext_vector_type(8)))  __bf16 v8bf;
typedef __attribute__((ext_vector_type(8)))  float  v8f;

#define WSZ 7
#define NH 12
#define DIM 384
#define HDD 32
#define NTOK 49
#define MTOK 64          // token dim padded to 4 WMMA tiles
#define NWGRID 15        // 105/7 windows per axis
#define NWIN_PER_B 225
#define BATCH 16
#define NWIN (BATCH*NWIN_PER_B)   // 3600
#define K3 (3*DIM)       // 1152
#define NTILES (K3/16)   // 72 output col tiles
#define KSTEPS (DIM/32)  // 12 K-steps of 32
#define IMGHW 100

__device__ __forceinline__ v8f wmma_bf16(v16bf a, v16bf b, v8f c) {
  // (neg_a, A, neg_b, B, c_mod, C, reuse_a, reuse_b)
  return __builtin_amdgcn_wmma_f32_16x16x32_bf16(false, a, false, b, (short)0, c,
                                                 false, false);
}

// Build a 16-element bf16 fragment from two contiguous 16-byte chunks.
// Matches the CDNA5 16-bit A/B VGPR layout: elements 0..7 = K {c..c+7},
// elements 8..15 = K {c+16..c+23}, with c = 8*(lane/16).
__device__ __forceinline__ v16bf ldfrag(const __bf16* p0, const __bf16* p1) {
  v8bf lo = *(const v8bf*)p0;
  v8bf hi = *(const v8bf*)p1;
  v16bf r;
#pragma unroll
  for (int i = 0; i < 8; ++i) { r[i] = lo[i]; r[i + 8] = hi[i]; }
  return r;
}

// ---------------------------------------------------------------------------
// Kernel 1: repack w_qkv (384 x 1152, f32 row-major) into bf16 B-fragment
// order: wB[nt][ks][lane][e], so a GEMM B fragment is 2 x global_load_b128.
// ---------------------------------------------------------------------------
__global__ __launch_bounds__(256) void prep_wB(const float* __restrict__ w,
                                               unsigned short* __restrict__ wBp) {
  int id = blockIdx.x * 256 + threadIdx.x;
  if (id >= NTILES * KSTEPS * 32 * 16) return;
  int e    = id & 15;
  int lane = (id >> 4) & 31;
  int ks   = (id >> 9) % KSTEPS;
  int nt   = id / (512 * KSTEPS);
  int n = lane & 15, half = lane >> 4;
  int k = ks * 32 + 8 * half + (e & 7) + ((e >> 3) << 4);
  ((__bf16*)wBp)[id] = (__bf16)w[(size_t)k * K3 + nt * 16 + n];
}

// ---------------------------------------------------------------------------
// Kernel 2: per-window QKV projection. One 256-thread (8-wave) block per
// window. (64 x 384) @ (384 x 1152) + bias, output bf16:
//   Q,K : [win][head][tok(64)][hd(32)]      (A / B-frag friendly for QK^T)
//   Vt  : [win][head][hd(32)][tok(64)]      (B-frag friendly for P@V)
// Padded / out-of-image tokens get x=0 -> qkv = bias, same as the reference.
// ---------------------------------------------------------------------------
__global__ __launch_bounds__(256) void qkv_win_gemm(
    const float* __restrict__ x, const unsigned short* __restrict__ wBp,
    const float* __restrict__ bqkv,
    unsigned short* __restrict__ Qp, unsigned short* __restrict__ Kp,
    unsigned short* __restrict__ Vtp) {
  __shared__ __bf16 xA[MTOK * DIM];  // 48 KB
  const __bf16* wB = (const __bf16*)wBp;
  __bf16* Qh = (__bf16*)Qp;
  __bf16* Kh = (__bf16*)Kp;
  __bf16* Vt = (__bf16*)Vtp;

  int bw = blockIdx.x;
  int b = bw / NWIN_PER_B, win = bw % NWIN_PER_B;
  int wy = win / NWGRID, wx = win % NWGRID;

  // Stage the window's x tile into LDS as bf16 (invalid rows -> 0).
  for (int p = threadIdx.x; p < MTOK * (DIM / 2); p += 256) {
    int row = p / (DIM / 2), cp = p % (DIM / 2);
    float v0 = 0.f, v1 = 0.f;
    if (row < NTOK) {
      int gy = wy * WSZ + row / WSZ, gx = wx * WSZ + row % WSZ;
      if (gy < IMGHW && gx < IMGHW) {
        const float2 f2 =
            *(const float2*)(x + (((size_t)(b * IMGHW + gy) * IMGHW + gx) * DIM +
                                  cp * 2));
        v0 = f2.x; v1 = f2.y;
      }
    }
    xA[row * DIM + cp * 2]     = (__bf16)v0;
    xA[row * DIM + cp * 2 + 1] = (__bf16)v1;
  }
  __syncthreads();

  int wave = threadIdx.x >> 5, lane = threadIdx.x & 31;
  int n = lane & 15, half = lane >> 4, c0 = 8 * half;
  size_t base = (size_t)bw * (NH * MTOK * HDD);

  for (int nt = wave; nt < NTILES; nt += 8) {
    v8f acc[4] = {};
    const __bf16* bptr = wB + (size_t)nt * KSTEPS * 512 + lane * 16;
#pragma unroll 2
    for (int ks = 0; ks < KSTEPS; ++ks) {
      v16bf bf = ldfrag(bptr + ks * 512, bptr + ks * 512 + 8);
#pragma unroll
      for (int m = 0; m < 4; ++m) {
        const __bf16* ap = &xA[(m * 16 + n) * DIM + ks * 32 + c0];
        v16bf af = ldfrag(ap, ap + 16);
        acc[m] = wmma_bf16(af, bf, acc[m]);
      }
    }
    int col = nt * 16 + n;
    float bb = bqkv[col];
    int mat = col / DIM, rem = col % DIM, head = rem >> 5, hd = rem & 31;
#pragma unroll
    for (int m = 0; m < 4; ++m)
#pragma unroll
      for (int i = 0; i < 8; ++i) {
        int tok = m * 16 + i + 8 * half;       // D layout: row = i + 8*(lane/16)
        float v = acc[m][i] + bb;
        if (mat == 0)
          Qh[base + ((size_t)head * MTOK + tok) * HDD + hd] = (__bf16)v;
        else if (mat == 1)
          Kh[base + ((size_t)head * MTOK + tok) * HDD + hd] = (__bf16)v;
        else
          Vt[base + ((size_t)head * HDD + hd) * MTOK + tok] = (__bf16)v;
      }
  }
}

// ---------------------------------------------------------------------------
// Kernel 3: per-(window, head) attention. 2 waves per 64-thread block, one
// head per wave. Dynamic LDS per block: 2 * (64*64 f32 S + 64*64 bf16 P).
// ---------------------------------------------------------------------------
__global__ __launch_bounds__(64) void win_attention(
    const unsigned short* __restrict__ Qp, const unsigned short* __restrict__ Kp,
    const unsigned short* __restrict__ Vtp, const float* __restrict__ table,
    float* __restrict__ out) {
  extern __shared__ char smem[];
  int wave = threadIdx.x >> 5, lane = threadIdx.x & 31;
  float*  S = (float*)smem + (size_t)wave * MTOK * MTOK;
  __bf16* P = (__bf16*)(smem + 2 * MTOK * MTOK * 4) + (size_t)wave * MTOK * MTOK;

  int bw = blockIdx.x / 6, hg = blockIdx.x % 6;
  int head = hg * 2 + wave;
  int b = bw / NWIN_PER_B, win = bw % NWIN_PER_B;
  int wy = win / NWGRID, wx = win % NWGRID;

  size_t hbase = (size_t)bw * (NH * MTOK * HDD) + (size_t)head * (MTOK * HDD);
  const __bf16* Q = (const __bf16*)Qp + hbase;   // [tok64][32]
  const __bf16* K = (const __bf16*)Kp + hbase;   // [tok64][32]
  const __bf16* V = (const __bf16*)Vtp + hbase;  // [32][tok64]

  int n = lane & 15, half = lane >> 4, c0 = 8 * half;

  // S = scale * Q @ K^T  (K rows are B-fragment columns)
  v16bf kf[4];
#pragma unroll
  for (int t = 0; t < 4; ++t) {
    const __bf16* kp = K + (size_t)(t * 16 + n) * HDD + c0;
    kf[t] = ldfrag(kp, kp + 16);
  }
  const float scale = 0.17677669529663689f;  // 32^-0.5
#pragma unroll
  for (int m = 0; m < 4; ++m) {
    const __bf16* qp = Q + (size_t)(m * 16 + n) * HDD + c0;
    v16bf qf = ldfrag(qp, qp + 16);
#pragma unroll
    for (int t = 0; t < 4; ++t) {
      v8f d = {};
      d = wmma_bf16(qf, kf[t], d);
#pragma unroll
      for (int i = 0; i < 8; ++i)
        S[(m * 16 + i + 8 * half) * MTOK + t * 16 + n] = d[i] * scale;
    }
  }

  // Row softmax with padding mask + relative position bias. Lane r owns
  // rows r and r+32 (all DS ops are same-wave, kept in order by hardware).
  for (int rr = 0; rr < 2; ++rr) {
    int row = lane + rr * 32;
    float*  Srow = S + row * MTOK;
    __bf16* Prow = P + row * MTOK;
    if (row < NTOK) {
      int qy = row / WSZ, qx = row % WSZ;
      float mx = -3.0e38f;
      for (int j = 0; j < NTOK; ++j) {
        int ky = j / WSZ, kx = j % WSZ;
        bool msk = (wy * WSZ + ky >= IMGHW) || (wx * WSZ + kx >= IMGHW);
        float bias =
            table[(size_t)((qy - ky + WSZ - 1) * 13 + (qx - kx + WSZ - 1)) * NH +
                  head];
        float v = (msk ? -10000.0f : Srow[j]) + bias;
        Srow[j] = v;
        mx = fmaxf(mx, v);
      }
      float sum = 0.f;
      for (int j = 0; j < NTOK; ++j) {
        float e = __expf(Srow[j] - mx);
        Srow[j] = e;
        sum += e;
      }
      float inv = 1.0f / sum;
      for (int j = 0; j < MTOK; ++j)
        Prow[j] = (__bf16)((j < NTOK) ? Srow[j] * inv : 0.f);
    } else {
      for (int j = 0; j < MTOK; ++j) Prow[j] = (__bf16)0.f;  // padded A rows
    }
  }

  // O = P @ V  (K-dim = 64 tokens -> 2 WMMA K-steps; N-dim = 32 -> 2 tiles)
  v8f o[4][2] = {};
#pragma unroll
  for (int s = 0; s < 2; ++s) {
    v16bf vb[2];
#pragma unroll
    for (int t = 0; t < 2; ++t) {
      const __bf16* vp = V + (size_t)(t * 16 + n) * MTOK + s * 32 + c0;
      vb[t] = ldfrag(vp, vp + 16);
    }
#pragma unroll
    for (int m = 0; m < 4; ++m) {
      const __bf16* pp = P + (size_t)(m * 16 + n) * MTOK + s * 32 + c0;
      v16bf af = ldfrag(pp, pp + 16);
#pragma unroll
      for (int t = 0; t < 2; ++t) o[m][t] = wmma_bf16(af, vb[t], o[m][t]);
    }
  }

  // Scatter valid (cropped) outputs; lanes 0..15 per tile give 64B coalesced
  // stores per D element row. Every valid output element written exactly once.
#pragma unroll
  for (int m = 0; m < 4; ++m)
#pragma unroll
    for (int i = 0; i < 8; ++i) {
      int tok = m * 16 + i + 8 * half;
      if (tok < NTOK) {
        int gy = wy * WSZ + tok / WSZ, gx = wx * WSZ + tok % WSZ;
        if (gy < IMGHW && gx < IMGHW) {
          size_t off = ((size_t)(b * IMGHW + gy) * IMGHW + gx) * DIM + head * HDD;
#pragma unroll
          for (int t = 0; t < 2; ++t) out[off + t * 16 + n] = o[m][t][i];
        }
      }
    }
}

// ---------------------------------------------------------------------------
extern "C" void kernel_launch(void* const* d_in, const int* in_sizes, int n_in,
                              void* d_out, int out_size, void* d_ws, size_t ws_size,
                              hipStream_t stream) {
  const float* x     = (const float*)d_in[0];  // (16,100,100,384) f32
  const float* wqkv  = (const float*)d_in[1];  // (384,1152) f32
  const float* bqkv  = (const float*)d_in[2];  // (1152,) f32
  const float* table = (const float*)d_in[3];  // (13,13,12) f32
  float* out = (float*)d_out;                  // (16,100,100,384) f32

  // Workspace layout (bf16 halves): Q | K | Vt | wB  (~532 MB total)
  const size_t per = (size_t)NWIN * NH * MTOK * HDD;  // 88,473,600 elements
  unsigned short* Qp  = (unsigned short*)d_ws;
  unsigned short* Kp  = Qp + per;
  unsigned short* Vtp = Kp + per;
  unsigned short* wBp = Vtp + per;

  (void)in_sizes; (void)n_in; (void)out_size; (void)ws_size;

  prep_wB<<<(NTILES * KSTEPS * 32 * 16 + 255) / 256, 256, 0, stream>>>(wqkv, wBp);
  qkv_win_gemm<<<NWIN, 256, 0, stream>>>(x, wBp, bqkv, Qp, Kp, Vtp);
  win_attention<<<NWIN * 6, 64, 2 * MTOK * MTOK * 4 + 2 * MTOK * MTOK * 2,
                  stream>>>(Qp, Kp, Vtp, table, out);
}
